// _fasterRCNN_51754355917514
// MI455X (gfx1250) — matrix operand order repaired and checked by reading, not compiled, not run on hardware
//
#include <hip/hip_runtime.h>
#include <cstdint>

typedef __attribute__((ext_vector_type(16))) __bf16 v16bf;
typedef __attribute__((ext_vector_type(8)))  __bf16 v8bf;
typedef __attribute__((ext_vector_type(8)))  float  v8f;

static __device__ __forceinline__ __bf16 f2bf(float f) {
  union { float f; uint32_t u; } v; v.f = f;
  uint32_t r = v.u + 0x7FFFu + ((v.u >> 16) & 1u);
  unsigned short h = (unsigned short)(r >> 16);
  return __builtin_bit_cast(__bf16, h);
}

// Async copy 16 bytes global -> LDS (CDNA5 GLOBAL_LOAD_ASYNC_TO_LDS_B128, GV mode).
static __device__ __forceinline__ void async_copy_b128(uint32_t lds_off, const __bf16* gptr) {
  asm volatile("global_load_async_to_lds_b128 %0, %1, off"
               :: "v"(lds_off), "v"(gptr)
               : "memory");
}

static __device__ __forceinline__ uint32_t lds_offset(const void* p) {
  return (uint32_t)(uintptr_t)p;   // flat LDS address: low 32 bits are the LDS offset
}

// ---------------- row-normalize + bf16 conversion -----------------
__global__ void rownorm_kernel(const float* __restrict__ x,
                               __bf16* __restrict__ pbf,
                               __bf16* __restrict__ xn, int D) {
  int row = blockIdx.x;
  const float* xr = x + (size_t)row * D;
  float ss = 0.f;
  for (int c = threadIdx.x; c < D; c += blockDim.x) { float v = xr[c]; ss += v * v; }
  __shared__ float red[256];
  red[threadIdx.x] = ss; __syncthreads();
  for (int s = 128; s > 0; s >>= 1) {
    if ((int)threadIdx.x < s) red[threadIdx.x] += red[threadIdx.x + s];
    __syncthreads();
  }
  float scale = 1.0f / fmaxf(sqrtf(red[0]), 1e-6f);
  for (int c = threadIdx.x; c < D; c += blockDim.x) {
    float v = xr[c];
    pbf[(size_t)row * D + c] = f2bf(v);
    xn [(size_t)row * D + c] = f2bf(v * scale);
  }
}

// ------------- transpose f32 (R x C) -> bf16 (Cpad x R), zero pad -------------
__global__ void transpose_f32_to_bf16(const float* __restrict__ in,
                                      __bf16* __restrict__ out,
                                      int R, int C, int Cpad) {
  int idx = blockIdx.x * blockDim.x + threadIdx.x;
  int total = Cpad * R;
  if (idx >= total) return;
  int c = idx / R, r = idx - c * R;
  out[idx] = (c < C) ? f2bf(in[(size_t)r * C + c]) : f2bf(0.f);
}

// ---------------- WMMA bf16 GEMM, NT form: C = A(MxK) * B(NxK)^T ----------------
// B tile (64 x 64k) staged to LDS via async-to-LDS, double buffered.
// MODE 0: f32 out + bias, cols bounded by nreal
// MODE 2: bf16 out, zero diagonal (cosine adjacency)
// MODE 3: bf16 out, stored transposed (ldc = leading dim of transposed = M)
// MODE 4: bias + relu, bf16 out row-major
template<int MODE>
__global__ void __launch_bounds__(256)
wmma_gemm_nt(const __bf16* __restrict__ A, const __bf16* __restrict__ B,
             float* __restrict__ Cf, __bf16* __restrict__ Cb,
             const float* __restrict__ bias,
             int K, int lda, int ldb, int ldc, int nreal) {
  constexpr int LDS_LD = 72;  // bf16 elems/row: 144B = 36 banks -> conflict-free frag reads
  __shared__ alignas(16) __bf16 Bsm[2][64 * LDS_LD];

  const int tid  = threadIdx.x;
  const int lane = tid & 31;
  const int wave = tid >> 5;
  const int m0 = blockIdx.y * 128 + wave * 16;
  const int n0 = blockIdx.x * 64;
  const int row  = lane & 15;   // A: M row / B: N col  (per ISA fragment layout)
  const int half = lane >> 4;   // K half selector

  // async staging: 512 x 16B slots per 64x64 bf16 chunk, 2 per thread
  const int r0 = tid >> 3,          c0 = (tid & 7) * 8;
  const int r1 = (tid + 256) >> 3,  c1 = (tid & 7) * 8;
  const __bf16* gB0 = B + (size_t)(n0 + r0) * ldb + c0;
  const __bf16* gB1 = B + (size_t)(n0 + r1) * ldb + c1;
  const uint32_t l0 = lds_offset(&Bsm[0][r0 * LDS_LD + c0]);
  const uint32_t l1 = lds_offset(&Bsm[0][r1 * LDS_LD + c1]);
  const uint32_t bufStride = lds_offset(&Bsm[1][0]) - lds_offset(&Bsm[0][0]);

  const __bf16* aPtr = A + (size_t)(m0 + row) * lda + half * 8;

  const int nch = K >> 6;
  // prefetch chunk 0 into buffer 0
  async_copy_b128(l0, gB0);
  async_copy_b128(l1, gB1);

  v8f acc0 = {}; v8f acc1 = {}; v8f acc2 = {}; v8f acc3 = {};

  for (int ch = 0; ch < nch; ++ch) {
    const int buf = ch & 1;
    if (ch + 1 < nch) {
      const uint32_t lb = (buf ^ 1) ? bufStride : 0u;
      async_copy_b128(l0 + lb, gB0 + (size_t)(ch + 1) * 64);
      async_copy_b128(l1 + lb, gB1 + (size_t)(ch + 1) * 64);
      // current buffer's copies (issued 1 round ago) complete; newest 2 may remain in flight
      asm volatile("s_wait_asynccnt 0x2" ::: "memory");
    } else {
      asm volatile("s_wait_asynccnt 0x0" ::: "memory");
    }
    __syncthreads();   // all waves' copies for this buffer are now visible

    __builtin_prefetch(aPtr + (size_t)(ch + 2) * 64, 0, 1);  // global_prefetch_b8 on A stream

    const __bf16* bs = &Bsm[buf][0];
    const __bf16* ap = aPtr + (size_t)ch * 64;

    // hoist ALL fragment loads for this 64-K chunk: 2 A frags (global), 8 B frags (LDS)
    v8bf a00 = *reinterpret_cast<const v8bf*>(ap);
    v8bf a01 = *reinterpret_cast<const v8bf*>(ap + 16);
    v8bf a10 = *reinterpret_cast<const v8bf*>(ap + 32);
    v8bf a11 = *reinterpret_cast<const v8bf*>(ap + 48);
    v16bf af0 = __builtin_shufflevector(a00, a01, 0,1,2,3,4,5,6,7,8,9,10,11,12,13,14,15);
    v16bf af1 = __builtin_shufflevector(a10, a11, 0,1,2,3,4,5,6,7,8,9,10,11,12,13,14,15);

    const int k0 = half * 8;         // s = 0
    const int k1 = 32 + half * 8;    // s = 1
    #define LD_BFRAG(rr, kk) \
      __builtin_shufflevector( \
        *reinterpret_cast<const v8bf*>(bs + (rr) * LDS_LD + (kk)), \
        *reinterpret_cast<const v8bf*>(bs + (rr) * LDS_LD + (kk) + 16), \
        0,1,2,3,4,5,6,7,8,9,10,11,12,13,14,15)
    v16bf b00 = LD_BFRAG(row +  0, k0);
    v16bf b01 = LD_BFRAG(row + 16, k0);
    v16bf b02 = LD_BFRAG(row + 32, k0);
    v16bf b03 = LD_BFRAG(row + 48, k0);
    v16bf b10 = LD_BFRAG(row +  0, k1);
    v16bf b11 = LD_BFRAG(row + 16, k1);
    v16bf b12 = LD_BFRAG(row + 32, k1);
    v16bf b13 = LD_BFRAG(row + 48, k1);
    #undef LD_BFRAG

    acc0 = __builtin_amdgcn_wmma_f32_16x16x32_bf16(false, af0, false, b00, (short)0, acc0, false, false);
    acc1 = __builtin_amdgcn_wmma_f32_16x16x32_bf16(false, af0, false, b01, (short)0, acc1, false, false);
    acc2 = __builtin_amdgcn_wmma_f32_16x16x32_bf16(false, af0, false, b02, (short)0, acc2, false, false);
    acc3 = __builtin_amdgcn_wmma_f32_16x16x32_bf16(false, af0, false, b03, (short)0, acc3, false, false);
    acc0 = __builtin_amdgcn_wmma_f32_16x16x32_bf16(false, af1, false, b10, (short)0, acc0, false, false);
    acc1 = __builtin_amdgcn_wmma_f32_16x16x32_bf16(false, af1, false, b11, (short)0, acc1, false, false);
    acc2 = __builtin_amdgcn_wmma_f32_16x16x32_bf16(false, af1, false, b12, (short)0, acc2, false, false);
    acc3 = __builtin_amdgcn_wmma_f32_16x16x32_bf16(false, af1, false, b13, (short)0, acc3, false, false);

    __syncthreads();   // everyone done reading 'buf' before it is refilled next round
  }

  // C/D layout: lane<16 -> M=v, lane>=16 -> M=8+v ; N = lane&15
  v8f accs[4] = {acc0, acc1, acc2, acc3};
  const int orow = half * 8;
  const int ocol = row;
  #pragma unroll
  for (int t = 0; t < 4; ++t) {
    int gn = n0 + t * 16 + ocol;
    #pragma unroll
    for (int v = 0; v < 8; ++v) {
      int gm = m0 + orow + v;
      float val = accs[t][v];
      if constexpr (MODE == 0) {
        if (gn < nreal) Cf[(size_t)gm * ldc + gn] = val + (bias ? bias[gn] : 0.f);
      } else if constexpr (MODE == 2) {
        Cb[(size_t)gm * ldc + gn] = f2bf(gm == gn ? 0.f : val);
      } else if constexpr (MODE == 3) {
        Cb[(size_t)gn * ldc + gm] = f2bf(val);
      } else if constexpr (MODE == 4) {
        Cb[(size_t)gm * ldc + gn] = f2bf(fmaxf(val + bias[gn], 0.f));
      }
    }
  }
}

// ---------------- softmax over C=21 cols, one wave per row ----------------
__global__ void softmax_kernel(float* __restrict__ logits, int C) {
  int r = blockIdx.x;
  float v = ((int)threadIdx.x < C) ? logits[(size_t)r * C + threadIdx.x] : -3.4e38f;
  float m = v;
  for (int s = 16; s > 0; s >>= 1) m = fmaxf(m, __shfl_xor(m, s, 32));
  float e = ((int)threadIdx.x < C) ? __expf(v - m) : 0.f;
  float sum = e;
  for (int s = 16; s > 0; s >>= 1) sum += __shfl_xor(sum, s, 32);
  if ((int)threadIdx.x < C) logits[(size_t)r * C + threadIdx.x] = e / sum;
}

// ---------------- IoU adjacency bitmask ----------------
__global__ void iou_adj_kernel(const float* __restrict__ rois,
                               uint32_t* __restrict__ adj, int N) {
  int i = blockIdx.x;
  int wsz = N >> 5;
  float x1 = rois[i * 4 + 0], y1 = rois[i * 4 + 1];
  float x2 = rois[i * 4 + 2], y2 = rois[i * 4 + 3];
  for (int w = threadIdx.x; w < wsz; w += blockDim.x) {
    uint32_t bits = 0;
    int jb = w << 5;
    for (int b = 0; b < 32; ++b) {
      int j = jb + b;
      float iw = fminf(x2, rois[j * 4 + 2]) - fmaxf(x1, rois[j * 4 + 0]) + 1.f;
      float ih = fminf(y2, rois[j * 4 + 3]) - fmaxf(y1, rois[j * 4 + 1]) + 1.f;
      bits |= ((iw > 0.f) && (ih > 0.f) && (j != i)) ? (1u << b) : 0u;
    }
    adj[(size_t)i * wsz + w] = bits;
  }
}

__global__ void cc_init_kernel(int* __restrict__ l, int N) {
  int i = blockIdx.x * blockDim.x + threadIdx.x;
  if (i < N) l[i] = i;
}

__global__ void cc_min_kernel(const uint32_t* __restrict__ adj,
                              const int* __restrict__ lin,
                              int* __restrict__ lout, int N) {
  int i = blockIdx.x;
  int wsz = N >> 5;
  int m = 0x7FFFFFFF;
  for (int w = threadIdx.x; w < wsz; w += blockDim.x) {
    uint32_t bits = adj[(size_t)i * wsz + w];
    int jb = w << 5;
    while (bits) {
      int b = __builtin_ctz(bits);
      bits &= bits - 1;
      m = min(m, lin[jb + b]);
    }
  }
  __shared__ int red[128];
  red[threadIdx.x] = m; __syncthreads();
  for (int s = 64; s > 0; s >>= 1) {
    if ((int)threadIdx.x < s) red[threadIdx.x] = min(red[threadIdx.x], red[threadIdx.x + s]);
    __syncthreads();
  }
  if (threadIdx.x == 0) lout[i] = min(lin[i], red[0]);
}

__global__ void cc_jump_kernel(const int* __restrict__ lin, int* __restrict__ lout, int N) {
  int i = blockIdx.x * blockDim.x + threadIdx.x;
  if (i < N) lout[i] = lin[lin[i]];
}

__global__ void labels_out_kernel(const int* __restrict__ l, float* __restrict__ out, int N) {
  int i = blockIdx.x * blockDim.x + threadIdx.x;
  if (i < N) out[i] = (float)l[i];
}

// =======================================================================
extern "C" void kernel_launch(void* const* d_in, const int* in_sizes, int n_in,
                              void* d_out, int out_size, void* d_ws, size_t ws_size,
                              hipStream_t stream) {
  const float* rois   = (const float*)d_in[0];
  const float* pooled = (const float*)d_in[1];
  const float* Wg1    = (const float*)d_in[2];
  const float* bg1    = (const float*)d_in[3];
  const float* Wg2    = (const float*)d_in[4];
  const float* bg2    = (const float*)d_in[5];
  const float* Wbbox  = (const float*)d_in[6];
  const float* bbbox  = (const float*)d_in[7];
  const float* Wcls   = (const float*)d_in[8];
  const float* bcls   = (const float*)d_in[9];

  const int N = 4096, D = 1024, H = 512, C = 21, B4 = 84, B4P = 128, CP = 64;

  char* ws = (char*)d_ws; size_t off = 0;
  auto carve = [&](size_t bytes) -> char* {
    char* p = ws + off; off = (off + bytes + 255) & ~(size_t)255; return p;
  };
  __bf16*  pbf  = (__bf16*)carve((size_t)N * D * 2);
  __bf16*  xn   = (__bf16*)carve((size_t)N * D * 2);
  __bf16*  Abf  = (__bf16*)carve((size_t)N * N * 2);
  __bf16*  Wg1T = (__bf16*)carve((size_t)H * D * 2);
  __bf16*  Wg2T = (__bf16*)carve((size_t)H * H * 2);
  __bf16*  WbT  = (__bf16*)carve((size_t)B4P * H * 2);
  __bf16*  WcT  = (__bf16*)carve((size_t)CP * H * 2);
  __bf16*  P1T  = (__bf16*)carve((size_t)H * N * 2);
  __bf16*  H1   = (__bf16*)carve((size_t)N * H * 2);
  __bf16*  P2T  = (__bf16*)carve((size_t)H * N * 2);
  __bf16*  H2   = (__bf16*)carve((size_t)N * H * 2);
  uint32_t* adj = (uint32_t*)carve((size_t)N * (N / 32) * 4);
  int* labA     = (int*)carve((size_t)N * 4);
  int* labB     = (int*)carve((size_t)N * 4);

  float* outLabels = (float*)d_out;
  float* outCls    = outLabels + N;
  float* outBbox   = outCls + (size_t)N * C;

  // --- precision prep ---
  rownorm_kernel<<<N, 256, 0, stream>>>(pooled, pbf, xn, D);
  transpose_f32_to_bf16<<<(H * D + 255) / 256, 256, 0, stream>>>(Wg1,   Wg1T, D, H,  H);
  transpose_f32_to_bf16<<<(H * H + 255) / 256, 256, 0, stream>>>(Wg2,   Wg2T, H, H,  H);
  transpose_f32_to_bf16<<<(B4P * H + 255) / 256, 256, 0, stream>>>(Wbbox, WbT, H, B4, B4P);
  transpose_f32_to_bf16<<<(CP * H + 255) / 256, 256, 0, stream>>>(Wcls,  WcT, H, C,  CP);

  dim3 blk(256);
  // A = zero_diag(xn @ xn^T)            [4096x4096 bf16]
  wmma_gemm_nt<2><<<dim3(N / 64, N / 128), blk, 0, stream>>>(xn, xn, nullptr, Abf, nullptr, D, D, D, N, N);
  // P1^T = (pooled @ Wg1)^T             [512x4096 bf16]
  wmma_gemm_nt<3><<<dim3(H / 64, N / 128), blk, 0, stream>>>(pbf, Wg1T, nullptr, P1T, nullptr, D, D, D, N, H);
  // H1 = relu(A @ P1 + bg1)             [4096x512 bf16]
  wmma_gemm_nt<4><<<dim3(H / 64, N / 128), blk, 0, stream>>>(Abf, P1T, nullptr, H1, bg1, N, N, N, H, H);
  // P2^T = (H1 @ Wg2)^T                 [512x4096 bf16]
  wmma_gemm_nt<3><<<dim3(H / 64, N / 128), blk, 0, stream>>>(H1, Wg2T, nullptr, P2T, nullptr, H, H, H, N, H);
  // H2 = relu(A @ P2 + bg2)             [4096x512 bf16]
  wmma_gemm_nt<4><<<dim3(H / 64, N / 128), blk, 0, stream>>>(Abf, P2T, nullptr, H2, bg2, N, N, N, H, H);
  // bbox_pred = H2 @ W_bbox + b_bbox    [4096x84 f32, padded to 128 cols]
  wmma_gemm_nt<0><<<dim3(B4P / 64, N / 128), blk, 0, stream>>>(H2, WbT, outBbox, nullptr, bbbox, H, H, H, B4, B4);
  // cls logits = H2 @ W_cls + b_cls     [4096x21 f32, padded to 64 cols]
  wmma_gemm_nt<0><<<dim3(CP / 64, N / 128), blk, 0, stream>>>(H2, WcT, outCls, nullptr, bcls, H, H, H, C, C);
  softmax_kernel<<<N, 32, 0, stream>>>(outCls, C);

  // --- connected components over IoU graph ---
  iou_adj_kernel<<<N, 128, 0, stream>>>(rois, adj, N);
  cc_init_kernel<<<(N + 255) / 256, 256, 0, stream>>>(labA, N);
  for (int it = 0; it < 32; ++it) {
    cc_min_kernel<<<N, 128, 0, stream>>>(adj, labA, labB, N);
    cc_jump_kernel<<<(N + 255) / 256, 256, 0, stream>>>(labB, labA, N);
  }
  labels_out_kernel<<<(N + 255) / 256, 256, 0, stream>>>(labA, outLabels, N);
}
